// Classifier_42588895707507
// MI455X (gfx1250) — matrix operand order repaired and checked by reading, not compiled, not run on hardware
//
#include <hip/hip_runtime.h>
#include <hip/hip_bf16.h>
#include <math.h>

typedef __attribute__((ext_vector_type(16))) __bf16 v16bf;
typedef __attribute__((ext_vector_type(8)))  __bf16 v8bf;
typedef __attribute__((ext_vector_type(8)))  float  v8f;
typedef int v4i __attribute__((ext_vector_type(4)));
#define AS1 __attribute__((address_space(1)))
#define AS3 __attribute__((address_space(3)))

#define WMMA_BF16(a, b, c) \
  __builtin_amdgcn_wmma_f32_16x16x32_bf16(false, (a), false, (b), (short)0, (c), false, false)

// -------- async global->LDS copy (CDNA5 ASYNCcnt path), with safe fallback --------
#if defined(__has_builtin)
#if __has_builtin(__builtin_amdgcn_global_load_async_to_lds_b128)
#define HAVE_ASYNC_LDS 1
#endif
#if __has_builtin(__builtin_amdgcn_s_wait_asynccnt)
#define HAVE_ASYNC_WAIT 1
#endif
#endif

__device__ __forceinline__ void async_cp16(const void* g, void* l) {
#ifdef HAVE_ASYNC_LDS
  // AS(1) pointer bits == generic pointer bits for global memory.
  AS1 v4i* gp = (AS1 v4i*)(unsigned long long)(uintptr_t)g;
  // Generic shared pointer = {aperture_hi, lds_offset_lo32}; low 32 bits are the LDS address.
  AS3 v4i* lp = (AS3 v4i*)(unsigned int)(uintptr_t)l;
  __builtin_amdgcn_global_load_async_to_lds_b128(gp, lp, 0, 0);
#else
  *(int4*)l = *(const int4*)g;
#endif
}
__device__ __forceinline__ void async_wait0() {
#ifdef HAVE_ASYNC_LDS
#ifdef HAVE_ASYNC_WAIT
  __builtin_amdgcn_s_wait_asynccnt(0);
#else
  asm volatile("s_wait_asynccnt 0" ::: "memory");
#endif
#endif
}

// ---------------- fragment loaders (contiguous k = 16*(lane>>4) + element, same for A and B, wave32) ----------------
// A 16x32 bf16: lane m=lane&15; lanes0-15 hold K 0..15, lanes16-31 hold K 16..31
__device__ __forceinline__ v16bf ldfa_bf(const __bf16* p, int ld) {
  int lane = threadIdx.x & 31;
  const __bf16* r = p + (size_t)(lane & 15) * ld + ((lane >> 4) << 4);
  union { v16bf v; v8bf h[2]; } u;
  u.h[0] = *(const v8bf*)r;
  u.h[1] = *(const v8bf*)(r + 8);
  return u.v;
}
// B 32x16 (= Wt with W row-major NxK): lane n=lane&15; lanes0-15 K 0..15, lanes16-31 K 16..31
__device__ __forceinline__ v16bf ldfb_bf(const __bf16* p, int ld) {
  int lane = threadIdx.x & 31;
  const __bf16* r = p + (size_t)(lane & 15) * ld + ((lane >> 4) << 4);
  return *(const v16bf*)r;
}
__device__ __forceinline__ v16bf ldfa_cvt(const float* p, int ld) {
  int lane = threadIdx.x & 31;
  const float* r = p + (size_t)(lane & 15) * ld + ((lane >> 4) << 4);
  v16bf v;
#pragma unroll
  for (int i = 0; i < 8; ++i) v[i] = (__bf16)r[i];
#pragma unroll
  for (int i = 0; i < 8; ++i) v[8 + i] = (__bf16)r[8 + i];
  return v;
}
__device__ __forceinline__ v16bf ldfb_cvt(const float* p, int ld) {
  int lane = threadIdx.x & 31;
  const float* r = p + (size_t)(lane & 15) * ld + ((lane >> 4) << 4);
  v16bf v;
#pragma unroll
  for (int i = 0; i < 16; ++i) v[i] = (__bf16)r[i];
  return v;
}

__device__ __forceinline__ unsigned short f32_to_f16b(float x) {
  _Float16 h = (_Float16)x; return __builtin_bit_cast(unsigned short, h);
}
__device__ __forceinline__ float f16b_to_f32(unsigned short b) {
  _Float16 h = __builtin_bit_cast(_Float16, b); return (float)h;
}
__device__ __forceinline__ unsigned short f32_to_bf16b(float x) {
  __bf16 h = (__bf16)x; return __builtin_bit_cast(unsigned short, h);
}

// ---------------- utility kernels ----------------
__global__ void f32_to_bf16_k(const float* __restrict__ in, __bf16* __restrict__ out, size_t n) {
  size_t i = (size_t)blockIdx.x * blockDim.x + threadIdx.x;
  if (i < n) out[i] = (__bf16)in[i];
}

__global__ void scan_init_k(float* hf, __bf16* hb, unsigned int* ctr) {
  int i = blockIdx.x * blockDim.x + threadIdx.x;
  if (i < 64 * 512) { hf[i] = 0.f; hb[i] = (__bf16)0.f; }
  if (i == 0) *ctr = 0u;
}

// ---------------- GEMM: C[Mx N] f32 = A[Mx512]bf16 . W[Nx512]bf16^T + bias ----------------
// K fixed at 512. Block = 256 thr (8 waves), tile 64x64. W tile staged in LDS via async copy.
__global__ __launch_bounds__(256) void gemm_bf16(const __bf16* __restrict__ A,
                                                 const __bf16* __restrict__ W,
                                                 const float* __restrict__ bias,
                                                 float* __restrict__ C, int N) {
  constexpr int K = 512;
  __shared__ __align__(128) __bf16 wl[64 * K];  // 64 KB
  int bm = blockIdx.x, bn = blockIdx.y;
  int tid = threadIdx.x;
  int w = tid >> 5, lane = tid & 31;
  int mt = w & 3, ntb = (w >> 2) * 2;

  // stage contiguous 64x512 W tile into LDS (async-to-LDS, 16B per op)
  {
    const __bf16* Wt = W + (size_t)(bn * 64) * K;
#pragma unroll
    for (int e = 0; e < 64 * K; e += 256 * 8)
      async_cp16(Wt + e + tid * 8, wl + e + tid * 8);
    async_wait0();
  }
  __syncthreads();

  const __bf16* Ap = A + (size_t)(bm * 64 + mt * 16) * K;
  const __bf16* L0 = wl + (size_t)(ntb * 16) * K;
  const __bf16* L1 = L0 + (size_t)16 * K;
  v8f c0 = {}, c1 = {};
  for (int k0 = 0; k0 < K; k0 += 32) {
    v16bf a  = ldfa_bf(Ap + k0, K);
    v16bf b0 = ldfb_bf(L0 + k0, K);
    v16bf b1 = ldfb_bf(L1 + k0, K);
    c0 = WMMA_BF16(a, b0, c0);
    c1 = WMMA_BF16(a, b1, c1);
  }
  int n0 = bn * 64 + ntb * 16 + (lane & 15);
  int mo = (lane >> 4) * 8;
  float* Cr = C + (size_t)(bm * 64 + mt * 16 + mo) * N;
  float bv0 = bias ? bias[n0] : 0.f;
  float bv1 = bias ? bias[n0 + 16] : 0.f;
#pragma unroll
  for (int i = 0; i < 8; ++i) {
    Cr[(size_t)i * N + n0]      = c0[i] + bv0;
    Cr[(size_t)i * N + n0 + 16] = c1[i] + bv1;
  }
}

// ---------------- persistent GRU scan ----------------
// 16 blocks x 256 thr. Block bk owns h columns bk*32..bk*32+31 (96 Whh rows).
// Whh slice (96x512 bf16 = 96KB) staged ONCE in LDS (time-invariant), then per t:
// gh = h(64x512,bf16) . Whh_slice^T via WMMA -> LDS, gates, grid barrier.
__global__ __launch_bounds__(256) void gru_scan_k(const float* __restrict__ gi,
                                                  const __bf16* __restrict__ Whh,
                                                  const float* __restrict__ bhh,
                                                  const float* __restrict__ xin,
                                                  float* __restrict__ xout,
                                                  float* __restrict__ hf,
                                                  __bf16* __restrict__ hb,
                                                  unsigned int* __restrict__ ctr) {
  __shared__ __align__(128) __bf16 wl[96 * 512];  // 96 KB
  __shared__ float ghs[64][96];                    // 24 KB
  const int bk = blockIdx.x;
  const int tid = threadIdx.x;
  const int w = tid >> 5, lane = tid & 31;
  const int mt = w & 3;             // batch tile 0..3
  const int ntb = (w >> 2) * 3;     // 3 gh-column tiles per wave

  // stage 6 contiguous 16x512 groups of the Whh slice (ordered by ntile)
  {
#pragma unroll
    for (int grp = 0; grp < 6; ++grp) {
      int g = grp >> 1, wi = grp & 1;
      const __bf16* src = Whh + (size_t)(g * 512 + bk * 32 + wi * 16) * 512;
      __bf16* dst = wl + (size_t)grp * 16 * 512;
#pragma unroll
      for (int e = 0; e < 16 * 512; e += 256 * 8)
        async_cp16(src + e + tid * 8, dst + e + tid * 8);
    }
    async_wait0();
  }
  __syncthreads();

  for (int t = 0; t < 256; ++t) {
    v8f acc[3] = {{}, {}, {}};
    const __bf16* hrow = hb + (size_t)(mt * 16) * 512;
    for (int k0 = 0; k0 < 512; k0 += 32) {
      v16bf a = ldfa_bf(hrow + k0, 512);
#pragma unroll
      for (int j = 0; j < 3; ++j) {
        v16bf b = ldfb_bf(wl + (size_t)(ntb + j) * 16 * 512 + k0, 512);
        acc[j] = WMMA_BF16(a, b, acc[j]);
      }
    }
    {
      int n = lane & 15, mo = (lane >> 4) * 8;
#pragma unroll
      for (int j = 0; j < 3; ++j)
#pragma unroll
        for (int i = 0; i < 8; ++i)
          ghs[mt * 16 + mo + i][(ntb + j) * 16 + n] = acc[j][i];
    }
    __syncthreads();
    // gates: 64 batches x 32 columns
    for (int e = tid; e < 2048; e += 256) {
      int b = e >> 5, c = e & 31;
      int col = bk * 32 + c;
      size_t gir = ((size_t)b * 256 + t) * 1536;
      float ir = gi[gir + col];
      float iz = gi[gir + 512 + col];
      float in_ = gi[gir + 1024 + col];
      float hr = ghs[b][c]      + bhh[col];
      float hz = ghs[b][32 + c] + bhh[512 + col];
      float hn = ghs[b][64 + c] + bhh[1024 + col];
      float r = 1.f / (1.f + __expf(-(ir + hr)));
      float z = 1.f / (1.f + __expf(-(iz + hz)));
      float nn = tanhf(in_ + r * hn);
      size_t hi = (size_t)b * 512 + col;
      float hold = hf[hi];
      float hnew = (1.f - z) * nn + z * hold;
      hf[hi] = hnew;
      hb[hi] = (__bf16)hnew;
      size_t xr = ((size_t)b * 256 + t) * 512 + col;
      xout[xr] = hnew + xin[xr];
    }
    __syncthreads();
    // grid barrier (monotonic counter; 16 blocks)
    __threadfence();
    if (tid == 0) {
      __hip_atomic_fetch_add(ctr, 1u, __ATOMIC_ACQ_REL, __HIP_MEMORY_SCOPE_AGENT);
      unsigned int target = 16u * (unsigned)(t + 1);
      while (__hip_atomic_load(ctr, __ATOMIC_ACQUIRE, __HIP_MEMORY_SCOPE_AGENT) < target)
        __builtin_amdgcn_s_sleep(4);
    }
    __syncthreads();
    __threadfence();
  }
}

// ---------------- attention: block per (query-chunk, head, batch) ----------------
// qkv rows (b*256+s), cols [q:0..511 | k:512..1023 | v:1024..1535]; HD=64, NH=8
__global__ __launch_bounds__(256) void attn_k(const float* __restrict__ qkv,
                                              const int* __restrict__ mask,
                                              float* __restrict__ out) {
  __shared__ __align__(64) __bf16 vts[64][272];          // V^T, bf16
  __shared__ __align__(64) unsigned short ps[128][272];  // scores f16 -> P bf16 (in place)
  __shared__ int mk[256];
  __shared__ float rs[128];
  const int qc = blockIdx.x;  // 0..1 (128-query chunk)
  const int h  = blockIdx.y;  // 0..7
  const int b  = blockIdx.z;  // 0..63
  const int tid = threadIdx.x;
  const int w = tid >> 5, lane = tid & 31;

  for (int i = tid; i < 64 * 256; i += 256) {
    int d = i >> 8, key = i & 255;
    vts[d][key] = (__bf16)qkv[((size_t)b * 256 + key) * 1536 + 1024 + h * 64 + d];
  }
  if (tid < 256) mk[tid] = mask[b * 256 + tid];
  __syncthreads();

  // scores for this wave's 16 query rows
  const int q0 = qc * 128 + w * 16;
  const float* qp = qkv + ((size_t)b * 256 + q0) * 1536 + h * 64;
  for (int j = 0; j < 16; ++j) {
    v8f s = {};
    const float* kp = qkv + ((size_t)b * 256 + j * 16) * 1536 + 512 + h * 64;
#pragma unroll
    for (int k0 = 0; k0 < 64; k0 += 32) {
      v16bf a  = ldfa_cvt(qp + k0, 1536);
      v16bf bb = ldfb_cvt(kp + k0, 1536);
      s = WMMA_BF16(a, bb, s);
    }
    int n = lane & 15, mo = (lane >> 4) * 8;
    float mval = (mk[j * 16 + n] == 0) ? -1e9f : 0.f;
#pragma unroll
    for (int i = 0; i < 8; ++i)
      ps[w * 16 + mo + i][j * 16 + n] = f32_to_f16b(s[i] * 0.125f + mval);
  }
  __syncthreads();

  // exact softmax, one row per thread (threads 0..127)
  if (tid < 128) {
    float mx = -3.4e38f;
    for (int c = 0; c < 256; ++c) mx = fmaxf(mx, f16b_to_f32(ps[tid][c]));
    float sum = 0.f;
    for (int c = 0; c < 256; ++c) {
      float p = __expf(f16b_to_f32(ps[tid][c]) - mx);
      sum += p;
      ps[tid][c] = f32_to_bf16b(p);  // unnormalized P, bf16
    }
    rs[tid] = sum;
  }
  __syncthreads();

  // out = P . V  (K = 256 keys, N = 64 dims)
  v8f o[4] = {{}, {}, {}, {}};
  const __bf16* Pp = (const __bf16*)&ps[w * 16][0];
  for (int k0 = 0; k0 < 256; k0 += 32) {
    v16bf a = ldfa_bf(Pp + k0, 272);
#pragma unroll
    for (int nt = 0; nt < 4; ++nt) {
      v16bf bb = ldfb_bf(&vts[nt * 16][0] + k0, 272);
      o[nt] = WMMA_BF16(a, bb, o[nt]);
    }
  }
  int n = lane & 15, mo = (lane >> 4) * 8;
#pragma unroll
  for (int nt = 0; nt < 4; ++nt)
#pragma unroll
    for (int i = 0; i < 8; ++i) {
      int row = qc * 128 + w * 16 + mo + i;
      float inv = 1.f / rs[w * 16 + mo + i];
      out[((size_t)b * 256 + row) * 512 + h * 64 + nt * 16 + n] = o[nt][i] * inv;
    }
}

// ---------------- masked mean pool ----------------
__global__ __launch_bounds__(256) void pool_k(const float* __restrict__ y,
                                              const int* __restrict__ mask,
                                              float* __restrict__ out) {
  __shared__ int msk[256];
  __shared__ int effs;
  int b = blockIdx.x, tid = threadIdx.x;
  msk[tid] = mask[b * 256 + tid];
  __syncthreads();
  if (tid == 0) {
    int e = 0;
    for (int s = 0; s < 256; ++s) e += msk[s];
    effs = e;
  }
  __syncthreads();
  float inv = 1.f / (float)effs;  // lengths >= 1 so effs >= 1
  for (int col = tid; col < 512; col += 256) {
    float acc = 0.f;
    for (int s = 0; s < 256; ++s)
      if (msk[s]) acc += y[((size_t)b * 256 + s) * 512 + col];
    out[b * 512 + col] = acc * inv;
  }
}

// ---------------- host orchestration ----------------
extern "C" void kernel_launch(void* const* d_in, const int* in_sizes, int n_in,
                              void* d_out, int out_size, void* d_ws, size_t ws_size,
                              hipStream_t stream) {
  const float* x_in  = (const float*)d_in[0];
  const int*   mask  = (const int*)d_in[1];
  const float* Wih_f = (const float*)d_in[2];
  const float* Whh_f = (const float*)d_in[3];
  const float* bih_f = (const float*)d_in[4];
  const float* bhh_f = (const float*)d_in[5];
  const float* Wih_b = (const float*)d_in[6];
  const float* Whh_b = (const float*)d_in[7];
  const float* bih_b = (const float*)d_in[8];
  const float* bhh_b = (const float*)d_in[9];
  const float* Wqkv  = (const float*)d_in[10];
  const float* bqkv  = (const float*)d_in[11];
  const float* Wo    = (const float*)d_in[12];
  const float* bo    = (const float*)d_in[13];

  char* ws = (char*)d_ws;
  size_t off = 0;
  auto alloc = [&](size_t bytes) { size_t o = off; off += (bytes + 255) & ~(size_t)255; return o; };

  const size_t MR = 16384;  // B*S rows
  float*  gi     = (float*)(ws + alloc(MR * 1536 * 4));   // gi / qkv / y (reused)
  float*  xa     = (float*)(ws + alloc(MR * 512 * 4));    // activations fp32
  __bf16* xbf    = (__bf16*)(ws + alloc(MR * 512 * 2));   // bf16 staging of activations
  __bf16* wihf   = (__bf16*)(ws + alloc(2ull * 1536 * 512 * 2));
  __bf16* whhf   = (__bf16*)(ws + alloc(2ull * 1536 * 512 * 2));
  __bf16* wihb   = (__bf16*)(ws + alloc(2ull * 1536 * 512 * 2));
  __bf16* whhb   = (__bf16*)(ws + alloc(2ull * 1536 * 512 * 2));
  __bf16* wqkvb  = (__bf16*)(ws + alloc(1536ull * 512 * 2));
  __bf16* wob    = (__bf16*)(ws + alloc(512ull * 512 * 2));
  float*  hf     = (float*)(ws + alloc(64ull * 512 * 4));
  __bf16* hbb    = (__bf16*)(ws + alloc(64ull * 512 * 2));
  unsigned int* ctr = (unsigned int*)(ws + alloc(256));

  auto conv = [&](const float* src, __bf16* dst, size_t n) {
    f32_to_bf16_k<<<(unsigned)((n + 255) / 256), 256, 0, stream>>>(src, dst, n);
  };

  // weights -> bf16
  conv(Wih_f, wihf, 2ull * 1536 * 512);
  conv(Whh_f, whhf, 2ull * 1536 * 512);
  conv(Wih_b, wihb, 2ull * 1536 * 512);
  conv(Whh_b, whhb, 2ull * 1536 * 512);
  conv(Wqkv,  wqkvb, 1536ull * 512);
  conv(Wo,    wob,   512ull * 512);

  // 4 sequential GRUs: (layer0,f) (layer0,b) (layer1,f) (layer1,b)
  for (int g = 0; g < 4; ++g) {
    int layer = g >> 1, dir = g & 1;
    const __bf16* wih = (dir ? wihb : wihf) + (size_t)layer * 1536 * 512;
    const __bf16* whh = (dir ? whhb : whhf) + (size_t)layer * 1536 * 512;
    const float*  bih = (dir ? bih_b : bih_f) + (size_t)layer * 1536;
    const float*  bhh = (dir ? bhh_b : bhh_f) + (size_t)layer * 1536;
    const float*  xsrc = (g == 0) ? x_in : xa;

    conv(xsrc, xbf, MR * 512);
    gemm_bf16<<<dim3(256, 24), 256, 0, stream>>>(xbf, wih, bih, gi, 1536);
    scan_init_k<<<128, 256, 0, stream>>>(hf, hbb, ctr);
    gru_scan_k<<<16, 256, 0, stream>>>(gi, whh, bhh, xsrc, xa, hf, hbb, ctr);
  }

  // MHA
  conv(xa, xbf, MR * 512);
  gemm_bf16<<<dim3(256, 24), 256, 0, stream>>>(xbf, wqkvb, bqkv, gi, 1536);  // qkv
  attn_k<<<dim3(2, 8, 64), 256, 0, stream>>>(gi, mask, xa);                   // attn -> xa
  conv(xa, xbf, MR * 512);
  gemm_bf16<<<dim3(256, 8), 256, 0, stream>>>(xbf, wob, bo, gi, 512);         // y -> gi

  // masked mean pool -> d_out (64 x 512 fp32)
  pool_k<<<64, 256, 0, stream>>>(gi, mask, (float*)d_out);
}
